// LCGNnet_13125420057193
// MI455X (gfx1250) — compile-verified, matrix operand order checked by reading, not run on hardware
//
#include <hip/hip_runtime.h>
#include <math.h>

// ---------------------------------------------------------------------------
// LCGN forward for MI455X (gfx1250, wave32).
//  - All GEMMs: v_wmma_f32_16x16x32_bf16 (bf16 inputs, fp32 accumulate).
//  - Weight matrices pre-converted to bf16 once per call; weight B-tiles are
//    DMA'd into LDS by the Tensor Data Mover (tensor_load_to_lds, TENSORcnt),
//    with D# LDS-padding reproducing the bank-conflict-free row stride.
//  - Activation-sourced B operands (attention GEMMs) use the fp32 VALU path.
// ---------------------------------------------------------------------------

typedef __attribute__((ext_vector_type(16))) __bf16 v16bf;
typedef __attribute__((ext_vector_type(8)))  float  v8f;
typedef __attribute__((ext_vector_type(4)))  unsigned int v4u;
typedef __attribute__((ext_vector_type(8)))  int v8i_;
typedef __attribute__((ext_vector_type(4)))  int v4i_;

#define B_    64
#define L_    30
#define N_    196
#define EMB_  300
#define ENC_  512
#define DCTX_ 512
#define DFEAT_ 2048
#define D_    256
#define G4_   1024
#define M2_   (B_ * N_)
#define NEGF  (-1e30f)

// ---------------- helpers ----------------

__device__ __forceinline__ unsigned short f2bf(float f) {
  union { float f; unsigned int u; } x; x.f = f;
  unsigned int u = x.u;
  u += 0x7FFFu + ((u >> 16) & 1u);           // round-to-nearest-even
  return (unsigned short)(u >> 16);
}

__device__ __forceinline__ float sigf(float x) { return 1.f / (1.f + __expf(-x)); }

__device__ __forceinline__ float blockReduceSum(float v, float* red) {
  int tid = threadIdx.x;
  red[tid] = v; __syncthreads();
  for (int s = 128; s > 0; s >>= 1) { if (tid < s) red[tid] += red[tid + s]; __syncthreads(); }
  float r = red[0]; __syncthreads(); return r;
}
__device__ __forceinline__ float blockReduceMax(float v, float* red) {
  int tid = threadIdx.x;
  red[tid] = v; __syncthreads();
  for (int s = 128; s > 0; s >>= 1) { if (tid < s) red[tid] = fmaxf(red[tid], red[tid + s]); __syncthreads(); }
  float r = red[0]; __syncthreads(); return r;
}

// ---------------- generic WMMA GEMM ----------------
// C[z] = act(outScale * A[z] @ op(B[z]) + bias + addMat), A fp32 -> bf16 staged.
// B16 != null : B is bf16 [N,K] row-major, tile loaded via Tensor Data Mover.
// else bT=1   : B fp32 [N,K] row-major.   bT=0 : B fp32 [K,N] row-major.

#define BM 128
#define BN 64
#define BK 32
#define LDSP 40   // padded LDS row stride (bf16 elems) = 80 bytes/row

union FragU { unsigned short u[16]; v16bf v; };

// Issue one TDM 2D tile load: BN x BK bf16 tile from a [*,K] bf16 tensor into
// LDS at ldsOff, padding each 64B row with 16B to land on the LDSP stride.
// D# packing per CDNA5 ISA ch.8 (group0 128b, group1 256b).
__device__ __forceinline__ void tdm_load_b_tile(const unsigned short* tile,
                                                unsigned int ldsOff,
                                                int K, int Krem, int Nrem)
{
  unsigned long long ga = (unsigned long long)(uintptr_t)tile;
  v4u g0;
  g0.x = 1u;                                              // count=1, user descriptor
  g0.y = ldsOff;                                          // lds_addr (bytes)
  g0.z = (unsigned int)(ga & 0xFFFFFFFFull);              // global_addr[31:0]
  g0.w = (unsigned int)((ga >> 32) & 0x1FFFFFFull)        // global_addr[56:32]
       | (2u << 30);                                      // type = 2 ("image")
  unsigned int td0 = (unsigned int)(Krem < 0 ? 0 : Krem); // tensor extent dim0
  unsigned int td1 = (unsigned int)(Nrem < 0 ? 0 : Nrem); // tensor extent dim1
  v8i_ g1;
  g1[0] = (1 << 16)        // data_size = 2 bytes
        | (1 << 20)        // pad_enable
        | (3 << 22)        // pad_interval: 16 DWORDs (64B) between pads
        | (3 << 25);       // pad_amount:   4 DWORDs (16B)
  g1[1] = (int)(td0 << 16);                    // tensor_dim0[15:0] @ bits 63:48
  g1[2] = (int)((td0 >> 16) | (td1 << 16));    // dim0 hi | dim1 lo
  g1[3] = (int)((td1 >> 16) | ((unsigned)BK << 16)); // dim1 hi | tile_dim0=32
  g1[4] = BN;                                  // tile_dim1 = 64, tile_dim2 = 0
  g1[5] = K;                                   // tensor_dim0_stride (elems)
  g1[6] = 0;
  g1[7] = 0;
  v4i_ gz4 = {0, 0, 0, 0};
  v8i_ gz8 = {0, 0, 0, 0, 0, 0, 0, 0};
  __builtin_amdgcn_tensor_load_to_lds(g0, g1, gz4, gz4, gz8, 0);
}

__global__ void __launch_bounds__(256) k_gemm(
    const float* __restrict__ A, const float* __restrict__ Bm,
    const unsigned short* __restrict__ B16,
    const float* __restrict__ bias, const float* __restrict__ addMat, int addLd,
    float* __restrict__ C, int M, int N, int K,
    int lda, int ldb, int ldc,
    long long sA, long long sB, long long sC,
    int bT, const float* __restrict__ rowScale, float outScale, int act)
{
  __shared__ unsigned short As[BM * LDSP];
  __shared__ unsigned short Bs[BN * LDSP];

  const int z = blockIdx.z;
  A  += (long long)z * sA;
  if (Bm) Bm += (long long)z * sB;
  C  += (long long)z * sC;

  const int rowBase = blockIdx.y * BM;
  const int colBase = blockIdx.x * BN;
  const int tid  = threadIdx.x;
  const int lane = tid & 31;
  const int wave = tid >> 5;
  const int wm = wave & 3;   // 4 row-groups of 32
  const int wn = wave >> 2;  // 2 col-groups of 32
  const bool useTdm = (B16 != nullptr);
  const unsigned int ldsOffB = (unsigned int)(uintptr_t)(&Bs[0]);

  v8f acc[2][2];
#pragma unroll
  for (int i = 0; i < 2; ++i)
#pragma unroll
    for (int j = 0; j < 2; ++j) {
      v8f zz = {0.f, 0.f, 0.f, 0.f, 0.f, 0.f, 0.f, 0.f};
      acc[i][j] = zz;
    }

  for (int k0 = 0; k0 < K; k0 += BK) {
    // ---- kick off async B tile DMA (weights, bf16) ----
    if (useTdm) {
      if (wave == 0)
        tdm_load_b_tile(B16 + (size_t)colBase * K + k0, ldsOffB, K, K - k0, N - colBase);
    }
    // ---- stage A tile (BM x BK) fp32 -> bf16 (overlaps TDM) ----
    for (int i = tid; i < BM * (BK / 4); i += 256) {
      int r  = i / (BK / 4);
      int kq = (i % (BK / 4)) * 4;
      int gr = rowBase + r, gk = k0 + kq;
      float v0 = 0.f, v1 = 0.f, v2 = 0.f, v3 = 0.f;
      if (gr < M) {
        if (gk + 3 < K) {
          const float4 t4 = *(const float4*)(A + (long long)gr * lda + gk);
          v0 = t4.x; v1 = t4.y; v2 = t4.z; v3 = t4.w;
        } else {
          if (gk     < K) v0 = A[(long long)gr * lda + gk];
          if (gk + 1 < K) v1 = A[(long long)gr * lda + gk + 1];
          if (gk + 2 < K) v2 = A[(long long)gr * lda + gk + 2];
          if (gk + 3 < K) v3 = A[(long long)gr * lda + gk + 3];
        }
        if (rowScale) { float s = rowScale[gr]; v0 *= s; v1 *= s; v2 *= s; v3 *= s; }
      }
      unsigned short* dst = &As[r * LDSP + kq];
      dst[0] = f2bf(v0); dst[1] = f2bf(v1); dst[2] = f2bf(v2); dst[3] = f2bf(v3);
    }
    // ---- stage B tile manually when source is fp32 activations ----
    if (!useTdm) {
      if (bT) {
        for (int i = tid; i < BN * (BK / 4); i += 256) {
          int n  = i / (BK / 4);
          int kq = (i % (BK / 4)) * 4;
          int gn = colBase + n, gk = k0 + kq;
          float v0 = 0.f, v1 = 0.f, v2 = 0.f, v3 = 0.f;
          if (gn < N) {
            if (gk + 3 < K) {
              const float4 t4 = *(const float4*)(Bm + (long long)gn * ldb + gk);
              v0 = t4.x; v1 = t4.y; v2 = t4.z; v3 = t4.w;
            } else {
              if (gk     < K) v0 = Bm[(long long)gn * ldb + gk];
              if (gk + 1 < K) v1 = Bm[(long long)gn * ldb + gk + 1];
              if (gk + 2 < K) v2 = Bm[(long long)gn * ldb + gk + 2];
              if (gk + 3 < K) v3 = Bm[(long long)gn * ldb + gk + 3];
            }
          }
          unsigned short* dst = &Bs[n * LDSP + kq];
          dst[0] = f2bf(v0); dst[1] = f2bf(v1); dst[2] = f2bf(v2); dst[3] = f2bf(v3);
        }
      } else {
        for (int i = tid; i < BN * BK; i += 256) {
          int n = i >> 5, k = i & 31;
          int gn = colBase + n, gk = k0 + k;
          float v = (gn < N && gk < K) ? Bm[(long long)gk * ldb + gn] : 0.f;
          Bs[n * LDSP + k] = f2bf(v);
        }
      }
    }
    // prefetch next A K-slab into the cache hierarchy (global_prefetch_b8)
    if (k0 + BK < K) {
      int pr = rowBase + (tid >> 1);
      if (pr < M) __builtin_prefetch(A + (long long)pr * lda + (k0 + BK), 0, 0);
    }
    // ---- close the async copy, publish LDS to the workgroup ----
    if (useTdm) {
      if (wave == 0) __builtin_amdgcn_s_wait_tensorcnt(0);
    }
    __syncthreads();

    // ---- build fragments per the gfx1250 bf16 WMMA VGPR layout ----
    FragU fa[2], fb[2];
    const int rsel = lane & 15;
    const int ks   = (lane >> 4) << 3;  // 0 or 8
#pragma unroll
    for (int t = 0; t < 2; ++t) {
      const unsigned short* ap = &As[(wm * 32 + t * 16 + rsel) * LDSP];
      const unsigned short* bp = &Bs[(wn * 32 + t * 16 + rsel) * LDSP];
#pragma unroll
      for (int e = 0; e < 8; ++e) {
        fa[t].u[e]     = ap[ks + e];
        fa[t].u[8 + e] = ap[ks + 16 + e];
        fb[t].u[e]     = bp[ks + e];
        fb[t].u[8 + e] = bp[ks + 16 + e];
      }
    }
#pragma unroll
    for (int i = 0; i < 2; ++i)
#pragma unroll
      for (int j = 0; j < 2; ++j)
        acc[i][j] = __builtin_amdgcn_wmma_f32_16x16x32_bf16(
            false, fa[i].v, false, fb[j].v, (short)0, acc[i][j], false, false);
    __syncthreads();
  }

  // ---- epilogue: C/D layout M = r + 8*(lane>>4), N = lane&15 ----
  const int mb = (lane >> 4) * 8;
  const int nc = lane & 15;
#pragma unroll
  for (int i = 0; i < 2; ++i) {
#pragma unroll
    for (int j = 0; j < 2; ++j) {
      int gn = colBase + wn * 32 + j * 16 + nc;
      float bv = (bias != nullptr && gn < N) ? bias[gn] : 0.f;
#pragma unroll
      for (int r = 0; r < 8; ++r) {
        int gm = rowBase + wm * 32 + i * 16 + mb + r;
        if (gm < M && gn < N) {
          float v = acc[i][j][r] * outScale + bv;
          if (addMat) v += addMat[(long long)gm * addLd + gn];
          if (act == 1) v = v > 0.f ? v : (__expf(v) - 1.f);  // ELU
          C[(long long)gm * ldc + gn] = v;
        }
      }
    }
  }
}

// ---------------- elementwise / small fused kernels ----------------

__global__ void k_cvt(const float* __restrict__ s, unsigned short* __restrict__ d, int n)
{
  int i = blockIdx.x * blockDim.x + threadIdx.x;
  if (i < n) d[i] = f2bf(s[i]);
}

__global__ void k_embed(const int* __restrict__ qIdx, const int* __restrict__ lens,
                        const float* __restrict__ emb, float* __restrict__ q,
                        float* __restrict__ qrev, int total)
{
  int idx = blockIdx.x * blockDim.x + threadIdx.x;
  if (idx >= total) return;
  int d = idx % EMB_;
  int t = (idx / EMB_) % L_;
  int b = idx / (EMB_ * L_);
  int tok = qIdx[b * L_ + t];
  q[idx] = (tok > 0) ? emb[(long long)(tok - 1) * EMB_ + d] : 0.f;
  int len = lens[b];
  float rv = 0.f;
  if (t < len) {
    int tok2 = qIdx[b * L_ + (len - 1 - t)];
    rv = (tok2 > 0) ? emb[(long long)(tok2 - 1) * EMB_ + d] : 0.f;
  }
  qrev[idx] = rv;
}

__global__ void k_lstm_update(const float* __restrict__ GF, const float* __restrict__ GB,
                              float* hF, float* cF, float* hB, float* cB,
                              float* outF, float* outBR, const int* __restrict__ lens, int t)
{
  int idx = blockIdx.x * blockDim.x + threadIdx.x;
  if (idx >= 2 * B_ * D_) return;
  int dir = idx / (B_ * D_);
  int rem = idx - dir * (B_ * D_);
  int b = rem / D_, j = rem % D_;
  const float* G = dir ? GB : GF;
  float* h = dir ? hB : hF;
  float* c = dir ? cB : cF;
  float* o = dir ? outBR : outF;
  float gi = G[b * G4_ + j];
  float gf = G[b * G4_ + D_ + j];
  float gg = G[b * G4_ + 2 * D_ + j];
  float go = G[b * G4_ + 3 * D_ + j];
  float cn = sigf(gf) * c[rem] + sigf(gi) * tanhf(gg);
  float hn = sigf(go) * tanhf(cn);
  bool valid = t < lens[b];
  if (valid) { c[rem] = cn; h[rem] = hn; }
  o[((long long)b * L_ + t) * D_ + j] = valid ? hn : 0.f;
}

__global__ void k_build_lstmout(const float* __restrict__ outF, const float* __restrict__ outBR,
                                const int* __restrict__ lens, float* __restrict__ lo, int total)
{
  int idx = blockIdx.x * blockDim.x + threadIdx.x;
  if (idx >= total) return;
  int d = idx % ENC_;
  int t = (idx / ENC_) % L_;
  int b = idx / (ENC_ * L_);
  if (d < D_) {
    lo[idx] = outF[((long long)b * L_ + t) * D_ + d];
  } else {
    int len = lens[b];
    float v = 0.f;
    if (t < len) v = outBR[((long long)b * L_ + (len - 1 - t)) * D_ + (d - D_)];
    lo[idx] = v;
  }
}

__global__ void k_build_vecQ(const float* hF, const float* hB, float* vq, int total)
{
  int idx = blockIdx.x * blockDim.x + threadIdx.x;
  if (idx >= total) return;
  int d = idx % ENC_, b = idx / ENC_;
  vq[idx] = (d < D_) ? hF[b * D_ + d] : hB[b * D_ + d - D_];
}

__global__ void k_rownorm(const float* __restrict__ img, float* __restrict__ invn)
{
  __shared__ float red[256];
  int row = blockIdx.x, tid = threadIdx.x;
  const float* r = img + (long long)row * DFEAT_;
  float ss = 0.f;
  for (int d = tid; d < DFEAT_; d += 256) { float v = r[d]; ss += v * v; }
  ss = blockReduceSum(ss, red);
  if (tid == 0) invn[row] = 1.f / fmaxf(sqrtf(ss), 1e-12f);
}

__global__ void k_fill_bcast(float* dst, const float* src, int width, int total)
{
  int idx = blockIdx.x * blockDim.x + threadIdx.x;
  if (idx < total) dst[idx] = src[idx % width];
}

__global__ void k_build_xj(const float* xl, const float* xc, const float* pl,
                           const float* pc, float* xj, int total)
{
  int idx = blockIdx.x * blockDim.x + threadIdx.x;
  if (idx >= total) return;
  int d = idx % (3 * DCTX_);
  long long row = idx / (3 * DCTX_);
  float v;
  if (d < DCTX_)          v = xl[row * DCTX_ + d];
  else if (d < 2 * DCTX_) v = xc[row * DCTX_ + d - DCTX_];
  else                    v = pl[row * DCTX_ + d - 2 * DCTX_] * pc[row * DCTX_ + d - 2 * DCTX_];
  xj[idx] = v;
}

__global__ void k_concat2(const float* A, const float* Bp, float* C, int w1, int w2, int total)
{
  int idx = blockIdx.x * blockDim.x + threadIdx.x;
  if (idx >= total) return;
  int w = w1 + w2;
  int d = idx % w;
  long long row = idx / w;
  C[idx] = (d < w1) ? A[row * w1 + d] : Bp[row * w2 + d - w1];
}

__global__ void k_mul_bcast(float* X, const float* s, int total)
{
  int idx = blockIdx.x * blockDim.x + threadIdx.x;
  if (idx >= total) return;
  int b = idx / (N_ * DCTX_);
  int d = idx % DCTX_;
  X[idx] *= s[b * DCTX_ + d];
}

__global__ void k_cmd_attend(const float* __restrict__ qcmd, const float* __restrict__ lo,
                             const float* __restrict__ cw, const float* __restrict__ cb,
                             const int* __restrict__ lens, float* __restrict__ cmd)
{
  __shared__ float red[256];
  __shared__ float att[32];
  int b = blockIdx.x, tid = threadIdx.x;
  int len = lens[b];
  const float* loB = lo + (long long)b * L_ * ENC_;
  const float* qc = qcmd + b * ENC_;
  for (int l = 0; l < L_; ++l) {
    float p = 0.f;
    for (int d = tid; d < ENC_; d += 256) p += qc[d] * loB[l * ENC_ + d] * cw[d];
    p = blockReduceSum(p, red);
    if (tid == 0) att[l] = (l < len) ? (p + cb[0]) : NEGF;
  }
  __syncthreads();
  if (tid == 0) {
    float mx = -3.4e38f;
    for (int l = 0; l < L_; ++l) mx = fmaxf(mx, att[l]);
    float s = 0.f;
    for (int l = 0; l < L_; ++l) { att[l] = __expf(att[l] - mx); s += att[l]; }
    float inv = 1.f / s;
    for (int l = 0; l < L_; ++l) att[l] *= inv;
  }
  __syncthreads();
  for (int d = tid; d < ENC_; d += 256) {
    float acc = 0.f;
    for (int l = 0; l < L_; ++l) acc += att[l] * loB[l * ENC_ + d];
    cmd[b * ENC_ + d] = acc;
  }
}

__global__ void k_ep_softmax(float* __restrict__ es, const int* __restrict__ entN)
{
  __shared__ float red[256];
  int n = blockIdx.x, b = blockIdx.y, tid = threadIdx.x;
  int e = entN[b];
  float* row = es + ((long long)b * N_ + n) * N_;
  bool nv = n < e;
  float mx = -3.4e38f;
  for (int m = tid; m < N_; m += 256) {
    float v = (nv && m < e) ? row[m] : NEGF;
    mx = fmaxf(mx, v);
  }
  mx = blockReduceMax(mx, red);
  float s = 0.f;
  for (int m = tid; m < N_; m += 256) {
    float v = (nv && m < e) ? row[m] : NEGF;
    s += __expf(v - mx);
  }
  s = blockReduceSum(s, red);
  float inv = 1.f / s;
  for (int m = tid; m < N_; m += 256) {
    float v = (nv && m < e) ? row[m] : NEGF;
    row[m] = __expf(v - mx) * inv;
  }
}

__global__ void k_sh_raw(const float* __restrict__ xout, const float* __restrict__ pq,
                         const float* __restrict__ w, const float* __restrict__ wb,
                         float* __restrict__ raw)
{
  __shared__ float red[256];
  int n = blockIdx.x, b = blockIdx.y, tid = threadIdx.x;
  const float* xo = xout + ((long long)b * N_ + n) * DCTX_;
  const float* p = pq + b * DCTX_;
  float ss = 0.f, dt = 0.f;
  for (int d = tid; d < DCTX_; d += 256) {
    float s = xo[d] * p[d];
    ss += s * s;
    dt += s * w[d];
  }
  ss = blockReduceSum(ss, red);
  dt = blockReduceSum(dt, red);
  if (tid == 0) raw[b * N_ + n] = dt / fmaxf(sqrtf(ss), 1e-12f) + wb[0];
}

__global__ void k_sh_attend(const float* __restrict__ raw, const float* __restrict__ xout,
                            const int* __restrict__ entN, float* __restrict__ xatt)
{
  __shared__ float red[256];
  __shared__ float att[200];
  int b = blockIdx.x, tid = threadIdx.x;
  int e = entN[b];
  const float* r = raw + b * N_;
  float mx = -3.4e38f;
  for (int n = tid; n < N_; n += 256) { float v = (n < e) ? r[n] : NEGF; mx = fmaxf(mx, v); }
  mx = blockReduceMax(mx, red);
  float s = 0.f;
  for (int n = tid; n < N_; n += 256) {
    float v = (n < e) ? r[n] : NEGF;
    float ev = __expf(v - mx);
    att[n] = ev;
    s += ev;
  }
  s = blockReduceSum(s, red);
  float inv = 1.f / s;
  const float* xo = xout + (long long)b * N_ * DCTX_;
  for (int d = tid; d < DCTX_; d += 256) {
    float acc = 0.f;
    for (int n = 0; n < N_; ++n) acc += att[n] * xo[n * DCTX_ + d];
    xatt[b * DCTX_ + d] = acc * inv;
  }
}

__global__ void k_feat(const float* xatt, const float* eQ, float* feat, int total)
{
  int idx = blockIdx.x * blockDim.x + threadIdx.x;
  if (idx >= total) return;
  int d = idx % (3 * DCTX_);
  int b = idx / (3 * DCTX_);
  float v;
  if (d < DCTX_)          v = xatt[b * DCTX_ + d];
  else if (d < 2 * DCTX_) v = eQ[b * DCTX_ + d - DCTX_];
  else                    v = xatt[b * DCTX_ + d - 2 * DCTX_] * eQ[b * DCTX_ + d - 2 * DCTX_];
  feat[idx] = v;
}

// ---------------- host ----------------

static void launch_gemm(hipStream_t st,
    const float* A, const float* Bm, const unsigned short* B16, const float* bias,
    const float* addMat, int addLd,
    float* C, int M, int N, int K,
    int lda, int ldb, int ldc,
    int batch, long long sA, long long sB, long long sC,
    int bT, const float* rowScale, float outScale, int act)
{
  dim3 grid((N + BN - 1) / BN, (M + BM - 1) / BM, batch);
  k_gemm<<<grid, 256, 0, st>>>(A, Bm, B16, bias, addMat, addLd, C, M, N, K,
                               lda, ldb, ldc, sA, sB, sC, bT, rowScale, outScale, act);
}

#define EW(kern, n, ...) kern<<<((n) + 255) / 256, 256, 0, stream>>>(__VA_ARGS__)

extern "C" void kernel_launch(void* const* d_in, const int* in_sizes, int n_in,
                              void* d_out, int out_size, void* d_ws, size_t ws_size,
                              hipStream_t stream)
{
  (void)in_sizes; (void)n_in; (void)out_size; (void)ws_size;
  const int*   qIdx     = (const int*)d_in[0];
  const int*   lens     = (const int*)d_in[1];
  const float* images   = (const float*)d_in[2];
  const int*   entN     = (const int*)d_in[3];
  const float* emb      = (const float*)d_in[4];
  const float* W_ih_f   = (const float*)d_in[5];
  const float* W_hh_f   = (const float*)d_in[6];
  const float* b_ih_f   = (const float*)d_in[7];
  const float* W_ih_b   = (const float*)d_in[8];
  const float* W_hh_b   = (const float*)d_in[9];
  const float* b_ih_b   = (const float*)d_in[10];
  const float* initKB_w = (const float*)d_in[11];
  const float* initKB_b = (const float*)d_in[12];
  const float* initMem  = (const float*)d_in[13];
  const float* qInput_w = (const float*)d_in[14];
  const float* qInput_b = (const float*)d_in[15];
  const float* qIt_w    = (const float*)d_in[16];
  const float* qIt_b    = (const float*)d_in[17];
  const float* cmd_w    = (const float*)d_in[18];
  const float* cmd_b    = (const float*)d_in[19];
  const float* ploc_w   = (const float*)d_in[20];
  const float* ploc_b   = (const float*)d_in[21];
  const float* pctx_w   = (const float*)d_in[22];
  const float* pctx_b   = (const float*)d_in[23];
  const float* q_w      = (const float*)d_in[24];
  const float* q_b      = (const float*)d_in[25];
  const float* k_w      = (const float*)d_in[26];
  const float* k_b      = (const float*)d_in[27];
  const float* v_w      = (const float*)d_in[28];
  const float* v_b      = (const float*)d_in[29];
  const float* pk_w     = (const float*)d_in[30];
  const float* pk_b     = (const float*)d_in[31];
  const float* pv_w     = (const float*)d_in[32];
  const float* pv_b     = (const float*)d_in[33];
  const float* mem_w    = (const float*)d_in[34];
  const float* mem_b    = (const float*)d_in[35];
  const float* comb_w   = (const float*)d_in[36];
  const float* comb_b   = (const float*)d_in[37];
  const float* sh_q_w   = (const float*)d_in[38];
  const float* sh_q_b   = (const float*)d_in[39];
  const float* outQ_w   = (const float*)d_in[40];
  const float* outQ_b   = (const float*)d_in[41];
  const float* cls1_w   = (const float*)d_in[42];
  const float* cls1_b   = (const float*)d_in[43];
  const float* shA_w    = (const float*)d_in[44];
  const float* shA_b    = (const float*)d_in[45];
  const float* cls2_w   = (const float*)d_in[46];
  const float* cls2_b   = (const float*)d_in[47];

  float* ws = (float*)d_ws;
  size_t off = 0;
  auto alloc  = [&](size_t n) { float* p = ws + off; off += n; return p; };
  auto allocH = [&](size_t n) { unsigned short* p = (unsigned short*)(ws + off); off += (n + 1) / 2; return p; };

  float* q     = alloc((size_t)B_ * L_ * EMB_);
  float* qrev  = alloc((size_t)B_ * L_ * EMB_);
  float* XihF  = alloc((size_t)B_ * L_ * G4_);
  float* XihB  = alloc((size_t)B_ * L_ * G4_);
  float* hF = alloc(B_ * D_); float* cF = alloc(B_ * D_);
  float* hB = alloc(B_ * D_); float* cB = alloc(B_ * D_);
  float* GF = alloc(B_ * G4_); float* GB = alloc(B_ * G4_);
  float* outF  = alloc((size_t)B_ * L_ * D_);
  float* outBR = alloc((size_t)B_ * L_ * D_);
  float* lstmOut = alloc((size_t)B_ * L_ * ENC_);
  float* vecQ  = alloc(B_ * ENC_);
  float* Ebuf  = alloc(B_ * ENC_);
  float* qcmd  = alloc(B_ * ENC_);
  float* cmd   = alloc(B_ * ENC_);
  float* pkc   = alloc(B_ * DCTX_);
  float* pvc   = alloc(B_ * DCTX_);
  float* invn  = alloc(M2_);
  float* xloc  = alloc((size_t)M2_ * DCTX_);
  float* xctxA = alloc((size_t)M2_ * DCTX_);
  float* xctxB = alloc((size_t)M2_ * DCTX_);
  float* plocB = alloc((size_t)M2_ * DCTX_);
  float* pctxB = alloc((size_t)M2_ * DCTX_);
  float* xj    = alloc((size_t)M2_ * 3 * DCTX_);
  float* qsB   = alloc((size_t)M2_ * DCTX_);
  float* kkB   = alloc((size_t)M2_ * DCTX_);
  float* vvB   = alloc((size_t)M2_ * DCTX_);
  float* esB   = alloc((size_t)B_ * N_ * N_);
  float* msg   = alloc((size_t)M2_ * DCTX_);
  float* cat2  = alloc((size_t)M2_ * 2 * DCTX_);
  float* xout  = alloc((size_t)M2_ * DCTX_);
  float* pq    = alloc(B_ * DCTX_);
  float* shraw = alloc(M2_);
  float* xatt  = alloc(B_ * DCTX_);
  float* eQ    = alloc(B_ * DCTX_);
  float* feat  = alloc((size_t)B_ * 3 * DCTX_);
  float* clsh  = alloc(B_ * 512);

  // ---- bf16 weight pool (converted once per call; L2 resident) ----
  unsigned short* Wihf16  = allocH((size_t)G4_ * EMB_);
  unsigned short* Wihb16  = allocH((size_t)G4_ * EMB_);
  unsigned short* Whhf16  = allocH((size_t)G4_ * D_);
  unsigned short* Whhb16  = allocH((size_t)G4_ * D_);
  unsigned short* initKB16= allocH((size_t)DCTX_ * DFEAT_);
  unsigned short* qIn16   = allocH((size_t)ENC_ * ENC_);
  unsigned short* qIt16   = allocH((size_t)4 * ENC_ * ENC_);
  unsigned short* ploc16  = allocH((size_t)DCTX_ * DCTX_);
  unsigned short* pctx16  = allocH((size_t)DCTX_ * DCTX_);
  unsigned short* qw16    = allocH((size_t)DCTX_ * 3 * DCTX_);
  unsigned short* kw16    = allocH((size_t)DCTX_ * 3 * DCTX_);
  unsigned short* vw16    = allocH((size_t)DCTX_ * 3 * DCTX_);
  unsigned short* pk16    = allocH((size_t)DCTX_ * DCTX_);
  unsigned short* pv16    = allocH((size_t)DCTX_ * DCTX_);
  unsigned short* mem16   = allocH((size_t)DCTX_ * 2 * DCTX_);
  unsigned short* comb16  = allocH((size_t)DCTX_ * 2 * DCTX_);
  unsigned short* shq16   = allocH((size_t)DCTX_ * ENC_);
  unsigned short* outQ16  = allocH((size_t)DCTX_ * ENC_);
  unsigned short* cls1_16 = allocH((size_t)512 * 3 * DCTX_);
  unsigned short* cls2_16 = allocH((size_t)1845 * 512);

  EW(k_cvt, G4_ * EMB_,        W_ih_f,   Wihf16,  G4_ * EMB_);
  EW(k_cvt, G4_ * EMB_,        W_ih_b,   Wihb16,  G4_ * EMB_);
  EW(k_cvt, G4_ * D_,          W_hh_f,   Whhf16,  G4_ * D_);
  EW(k_cvt, G4_ * D_,          W_hh_b,   Whhb16,  G4_ * D_);
  EW(k_cvt, DCTX_ * DFEAT_,    initKB_w, initKB16, DCTX_ * DFEAT_);
  EW(k_cvt, ENC_ * ENC_,       qInput_w, qIn16,   ENC_ * ENC_);
  EW(k_cvt, 4 * ENC_ * ENC_,   qIt_w,    qIt16,   4 * ENC_ * ENC_);
  EW(k_cvt, DCTX_ * DCTX_,     ploc_w,   ploc16,  DCTX_ * DCTX_);
  EW(k_cvt, DCTX_ * DCTX_,     pctx_w,   pctx16,  DCTX_ * DCTX_);
  EW(k_cvt, DCTX_ * 3 * DCTX_, q_w,      qw16,    DCTX_ * 3 * DCTX_);
  EW(k_cvt, DCTX_ * 3 * DCTX_, k_w,      kw16,    DCTX_ * 3 * DCTX_);
  EW(k_cvt, DCTX_ * 3 * DCTX_, v_w,      vw16,    DCTX_ * 3 * DCTX_);
  EW(k_cvt, DCTX_ * DCTX_,     pk_w,     pk16,    DCTX_ * DCTX_);
  EW(k_cvt, DCTX_ * DCTX_,     pv_w,     pv16,    DCTX_ * DCTX_);
  EW(k_cvt, DCTX_ * 2 * DCTX_, mem_w,    mem16,   DCTX_ * 2 * DCTX_);
  EW(k_cvt, DCTX_ * 2 * DCTX_, comb_w,   comb16,  DCTX_ * 2 * DCTX_);
  EW(k_cvt, DCTX_ * ENC_,      sh_q_w,   shq16,   DCTX_ * ENC_);
  EW(k_cvt, DCTX_ * ENC_,      outQ_w,   outQ16,  DCTX_ * ENC_);
  EW(k_cvt, 512 * 3 * DCTX_,   cls1_w,   cls1_16, 512 * 3 * DCTX_);
  EW(k_cvt, 1845 * 512,        cls2_w,   cls2_16, 1845 * 512);

  // ---- encoder: embedding + BiLSTM ----
  (void)hipMemsetAsync(hF, 0, B_ * D_ * sizeof(float), stream);
  (void)hipMemsetAsync(cF, 0, B_ * D_ * sizeof(float), stream);
  (void)hipMemsetAsync(hB, 0, B_ * D_ * sizeof(float), stream);
  (void)hipMemsetAsync(cB, 0, B_ * D_ * sizeof(float), stream);

  int nE = B_ * L_ * EMB_;
  EW(k_embed, nE, qIdx, lens, emb, q, qrev, nE);

  launch_gemm(stream, q,    nullptr, Wihf16, b_ih_f, nullptr, 0, XihF, B_ * L_, G4_, EMB_,
              EMB_, EMB_, G4_, 1, 0, 0, 0, 1, nullptr, 1.f, 0);
  launch_gemm(stream, qrev, nullptr, Wihb16, b_ih_b, nullptr, 0, XihB, B_ * L_, G4_, EMB_,
              EMB_, EMB_, G4_, 1, 0, 0, 0, 1, nullptr, 1.f, 0);

  for (int t = 0; t < L_; ++t) {
    launch_gemm(stream, hF, nullptr, Whhf16, nullptr, XihF + (long long)t * G4_, L_ * G4_,
                GF, B_, G4_, D_, D_, D_, G4_, 1, 0, 0, 0, 1, nullptr, 1.f, 0);
    launch_gemm(stream, hB, nullptr, Whhb16, nullptr, XihB + (long long)t * G4_, L_ * G4_,
                GB, B_, G4_, D_, D_, D_, G4_, 1, 0, 0, 0, 1, nullptr, 1.f, 0);
    int nU = 2 * B_ * D_;
    EW(k_lstm_update, nU, GF, GB, hF, cF, hB, cB, outF, outBR, lens, t);
  }
  int nLO = B_ * L_ * ENC_;
  EW(k_build_lstmout, nLO, outF, outBR, lens, lstmOut, nLO);
  int nV = B_ * ENC_;
  EW(k_build_vecQ, nV, hF, hB, vecQ, nV);

  // ---- LCGN init ----
  k_rownorm<<<M2_, 256, 0, stream>>>(images, invn);
  launch_gemm(stream, images, nullptr, initKB16, initKB_b, nullptr, 0, xloc, M2_, DCTX_, DFEAT_,
              DFEAT_, DFEAT_, DCTX_, 1, 0, 0, 0, 1, invn, 1.f, 0);
  int nC = M2_ * DCTX_;
  EW(k_fill_bcast, nC, xctxA, initMem, DCTX_, nC);
  launch_gemm(stream, vecQ, nullptr, qIn16, qInput_b, nullptr, 0, Ebuf, B_, ENC_, ENC_,
              ENC_, ENC_, ENC_, 1, 0, 0, 0, 1, nullptr, 1.f, 1 /*ELU(q_first)*/);

  const float invsq = 1.f / sqrtf((float)DCTX_);
  float* xc = xctxA;
  float* xn = xctxB;
  for (int t = 0; t < 4; ++t) {
    launch_gemm(stream, Ebuf, nullptr, qIt16 + (size_t)t * ENC_ * ENC_, qIt_b + t * ENC_,
                nullptr, 0, qcmd, B_, ENC_, ENC_, ENC_, ENC_, ENC_, 1, 0, 0, 0, 1, nullptr, 1.f, 0);
    k_cmd_attend<<<B_, 256, 0, stream>>>(qcmd, lstmOut, cmd_w, cmd_b, lens, cmd);

    launch_gemm(stream, xloc, nullptr, ploc16, ploc_b, nullptr, 0, plocB, M2_, DCTX_, DCTX_,
                DCTX_, DCTX_, DCTX_, 1, 0, 0, 0, 1, nullptr, 1.f, 0);
    launch_gemm(stream, xc,   nullptr, pctx16, pctx_b, nullptr, 0, pctxB, M2_, DCTX_, DCTX_,
                DCTX_, DCTX_, DCTX_, 1, 0, 0, 0, 1, nullptr, 1.f, 0);
    int nXJ = M2_ * 3 * DCTX_;
    EW(k_build_xj, nXJ, xloc, xc, plocB, pctxB, xj, nXJ);

    launch_gemm(stream, xj, nullptr, qw16, q_b, nullptr, 0, qsB, M2_, DCTX_, 3 * DCTX_,
                3 * DCTX_, 3 * DCTX_, DCTX_, 1, 0, 0, 0, 1, nullptr, 1.f, 0);
    launch_gemm(stream, xj, nullptr, kw16, k_b, nullptr, 0, kkB, M2_, DCTX_, 3 * DCTX_,
                3 * DCTX_, 3 * DCTX_, DCTX_, 1, 0, 0, 0, 1, nullptr, 1.f, 0);
    launch_gemm(stream, cmd, nullptr, pk16, pk_b, nullptr, 0, pkc, B_, DCTX_, DCTX_,
                DCTX_, DCTX_, DCTX_, 1, 0, 0, 0, 1, nullptr, 1.f, 0);
    EW(k_mul_bcast, nC, kkB, pkc, nC);
    launch_gemm(stream, xj, nullptr, vw16, v_b, nullptr, 0, vvB, M2_, DCTX_, 3 * DCTX_,
                3 * DCTX_, 3 * DCTX_, DCTX_, 1, 0, 0, 0, 1, nullptr, 1.f, 0);
    launch_gemm(stream, cmd, nullptr, pv16, pv_b, nullptr, 0, pvc, B_, DCTX_, DCTX_,
                DCTX_, DCTX_, DCTX_, 1, 0, 0, 0, 1, nullptr, 1.f, 0);
    EW(k_mul_bcast, nC, vvB, pvc, nC);

    // es[b] = qs[b] @ kk[b]^T * invsq   (batched, fp32 B, bT=1)
    launch_gemm(stream, qsB, kkB, nullptr, nullptr, nullptr, 0, esB, N_, N_, DCTX_,
                DCTX_, DCTX_, N_, B_, (long long)N_ * DCTX_, (long long)N_ * DCTX_,
                (long long)N_ * N_, 1, nullptr, invsq, 0);
    k_ep_softmax<<<dim3(N_, B_), 256, 0, stream>>>(esB, entN);
    // msg[b] = ep[b] @ vv[b]            (batched, fp32 B, bT=0)
    launch_gemm(stream, esB, vvB, nullptr, nullptr, nullptr, 0, msg, N_, DCTX_, N_,
                N_, DCTX_, DCTX_, B_, (long long)N_ * N_, (long long)N_ * DCTX_,
                (long long)N_ * DCTX_, 0, nullptr, 1.f, 0);

    int nC2 = M2_ * 2 * DCTX_;
    EW(k_concat2, nC2, xc, msg, cat2, DCTX_, DCTX_, nC2);
    launch_gemm(stream, cat2, nullptr, mem16, mem_b, nullptr, 0, xn, M2_, DCTX_, 2 * DCTX_,
                2 * DCTX_, 2 * DCTX_, DCTX_, 1, 0, 0, 0, 1, nullptr, 1.f, 0);
    float* tmp = xc; xc = xn; xn = tmp;
  }
  int nC2 = M2_ * 2 * DCTX_;
  EW(k_concat2, nC2, xloc, xc, cat2, DCTX_, DCTX_, nC2);
  launch_gemm(stream, cat2, nullptr, comb16, comb_b, nullptr, 0, xout, M2_, DCTX_, 2 * DCTX_,
              2 * DCTX_, 2 * DCTX_, DCTX_, 1, 0, 0, 0, 1, nullptr, 1.f, 0);

  // ---- SingleHop ----
  launch_gemm(stream, vecQ, nullptr, shq16, sh_q_b, nullptr, 0, pq, B_, DCTX_, ENC_,
              ENC_, ENC_, DCTX_, 1, 0, 0, 0, 1, nullptr, 1.f, 0);
  k_sh_raw<<<dim3(N_, B_), 256, 0, stream>>>(xout, pq, shA_w, shA_b, shraw);
  k_sh_attend<<<B_, 256, 0, stream>>>(shraw, xout, entN, xatt);

  // ---- classifier ----
  launch_gemm(stream, vecQ, nullptr, outQ16, outQ_b, nullptr, 0, eQ, B_, DCTX_, ENC_,
              ENC_, ENC_, DCTX_, 1, 0, 0, 0, 1, nullptr, 1.f, 0);
  int nF = B_ * 3 * DCTX_;
  EW(k_feat, nF, xatt, eQ, feat, nF);
  launch_gemm(stream, feat, nullptr, cls1_16, cls1_b, nullptr, 0, clsh, B_, 512, 3 * DCTX_,
              3 * DCTX_, 3 * DCTX_, 512, 1, 0, 0, 0, 1, nullptr, 1.f, 1 /*ELU*/);
  launch_gemm(stream, clsh, nullptr, cls2_16, cls2_b, nullptr, 0, (float*)d_out, B_, 1845, 512,
              512, 512, 1845, 1, 0, 0, 0, 1, nullptr, 1.f, 0);
}
